// SingleScaleRNGF_52862457480018
// MI455X (gfx1250) — matrix operand order repaired and checked by reading, not compiled, not run on hardware
//
#include <hip/hip_runtime.h>
#include <hip/hip_bf16.h>

typedef __attribute__((ext_vector_type(16))) _Float16 v16h;
typedef __attribute__((ext_vector_type(8)))  float    v8f;

#define VOL    160
#define TILE   16
#define TP     18            // tile + 1-voxel halo on each side
#define TPLANE (TP * TP)     // 324
#define TELEMS (TP * TP * TP)// 5832
#define EPS_F  1.1920928955078125e-07f  // float32 machine eps (2^-23)
#define NVOX   8192000.0f               // 2 * 160^3

// ws layout (floats):
//   [0..1] sum(mask)          per batch
//   [2..3] sum(norm_t * mask) per batch
//   [4..5] sum(norm_p * mask) per batch
//   [6]    sum(inner^2) over everything

// Sobel weight for output channel `row` (0=gx,1=gy,2=gz) at flattened
// neighbor index K = (dz*3+dy)*3+dx, dz,dy,dx in {0,1,2}. K>26 -> zero pad.
// SOBEL_X[z][y][x] = s[z]s[y]d[x]; SOBEL_Y = s[z]d[y]s[x]; SOBEL_Z = d[z]s[y]s[x]
__device__ __forceinline__ float sobel_w(int row, int K) {
    if (row > 2 || K > 26) return 0.0f;
    int dz = K / 9, rem = K - dz * 9, dy = rem / 3, dx = rem - dy * 3;
    const float s0[3] = {1.0f, 2.0f, 1.0f};
    const float d0[3] = {-1.0f, 0.0f, 1.0f};
    if (row == 0) return s0[dz] * s0[dy] * d0[dx];
    if (row == 1) return s0[dz] * d0[dy] * s0[dx];
    return d0[dz] * s0[dy] * s0[dx];
}

__device__ __forceinline__ int clampi(int v, int lo, int hi) {
    return v < lo ? lo : (v > hi ? hi : v);
}

__global__ void init_ws_kernel(float* ws) {
    if (threadIdx.x < 7) ws[threadIdx.x] = 0.0f;
}

__global__ __launch_bounds__(256)
void sobel_pass_kernel(const float* __restrict__ pred,
                       const float* __restrict__ targ,
                       const float* __restrict__ mask,
                       float* __restrict__ ws,
                       int mode)
{
    // Interleaved f16 tile: low 16 bits = target, high 16 bits = pred.
    __shared__ unsigned tPT[TELEMS];   // 23328 B
    __shared__ float    stage[TELEMS]; // 23328 B async-copy staging (f32)
    __shared__ float    red[4];

    const int tid = threadIdx.x;
    const int b  = blockIdx.z / 10;        // batch
    const int tz = blockIdx.z % 10;
    const int x0 = blockIdx.x * TILE;
    const int y0 = blockIdx.y * TILE;
    const int z0 = tz * TILE;
    const int base = b * (VOL * VOL * VOL);

    if (tid < 4) red[tid] = 0.0f;

    // speculative prefetch of the tile origin cachelines (global_prefetch_b8)
    __builtin_prefetch(pred + base + ((z0 * VOL + y0) * VOL + x0), 0, 1);
    __builtin_prefetch(targ + base + ((z0 * VOL + y0) * VOL + x0), 0, 1);

    // ---- Stage + pack both tiles. Async global->LDS (ASYNCcnt) with
    //      clamped addresses; zero-padding applied during f16 pack. ----
    #pragma unroll 1
    for (int img = 0; img < 2; ++img) {
        const float* src = (img == 0) ? targ : pred;
        for (int idx = tid; idx < TELEMS; idx += 256) {
            int lz = idx / TPLANE;
            int r  = idx - lz * TPLANE;
            int ly = r / TP;
            int lx = r - ly * TP;
            int gzc = clampi(z0 + lz - 1, 0, VOL - 1);
            int gyc = clampi(y0 + ly - 1, 0, VOL - 1);
            int gxc = clampi(x0 + lx - 1, 0, VOL - 1);
            const float* gaddr = src + (base + (gzc * VOL + gyc) * VOL + gxc);
            unsigned lds_off = (unsigned)(uintptr_t)&stage[idx];
            asm volatile("global_load_async_to_lds_b32 %0, %1, off"
                         :: "v"(lds_off), "v"(gaddr) : "memory");
        }
        asm volatile("s_wait_asynccnt 0x0" ::: "memory");
        __syncthreads();
        for (int idx = tid; idx < TELEMS; idx += 256) {
            int lz = idx / TPLANE;
            int r  = idx - lz * TPLANE;
            int ly = r / TP;
            int lx = r - ly * TP;
            int gz = z0 + lz - 1, gy = y0 + ly - 1, gx = x0 + lx - 1;
            bool in = ((unsigned)gz < (unsigned)VOL) &
                      ((unsigned)gy < (unsigned)VOL) &
                      ((unsigned)gx < (unsigned)VOL);
            float v = in ? stage[idx] : 0.0f;
            unsigned short h = __builtin_bit_cast(unsigned short, (_Float16)v);
            if (img == 0) tPT[idx] = (unsigned)h;            // target -> low half
            else          tPT[idx] |= ((unsigned)h) << 16;   // pred   -> high half
        }
        __syncthreads();  // also protects stage[] reuse between images
    }

    const int lane = tid & 31;
    const int wid  = tid >> 5;     // 8 wave32s per block
    const int g    = lane >> 4;    // lane half-group
    const int rn   = lane & 15;    // A row (output channel) / B column (x in strip)

    // A fragment: 16x32 fp16 weights, constant per lane.
    // A layout (ISA 7.12.2): lanes 0-15 V0..3=K0..7, V4..7=K16..23;
    //                        lanes 16-31 V0..3=K8..15, V4..7=K24..31.
    // B layout (K=32 x N=16): same K order as A,
    //                         element j of v16h = K (=8*g + j, j<8; 16+8*g+j-8, j>=8).
    v16h afrag;
    int bdel[16];
    #pragma unroll
    for (int j = 0; j < 16; ++j) {
        int Ka = (j < 8) ? (8 * g + j) : (16 + 8 * g + (j - 8));
        afrag[j] = (_Float16)sobel_w(rn, Ka);
        int Kb = Ka;
        int kk = (Kb > 26) ? 0 : Kb;   // K>=27 multiplies zero A columns
        int dz = kk / 9, rem = kk - dz * 9, dy = rem / 3, dx = rem - dy * 3;
        bdel[j] = dz * TPLANE + dy * TP + dx;
    }

    float aet = 0.0f, aep = 0.0f;
    if (mode == 1) {
        float rs  = __builtin_amdgcn_rcpf(ws[0 + b]);
        aet = ws[2 + b] * rs;   // eta = 1.0
        aep = ws[4 + b] * rs;
    }

    float accA = 0.0f, accB = 0.0f, accC = 0.0f;

    // Each wave handles one strip of 16 x-voxels at (zl, yl); 32 strips/wave.
    for (int s = wid; s < 256; s += 8) {
        int zl = s >> 4, yl = s & 15;
        int lbase = zl * TPLANE + yl * TP + rn;

        // One ds_load_b32 feeds both B fragments (interleaved halves).
        v16h bfT, bfP;
        #pragma unroll
        for (int j = 0; j < 16; ++j) {
            unsigned w = tPT[lbase + bdel[j]];
            bfT[j] = __builtin_bit_cast(_Float16, (unsigned short)(w & 0xffffu));
            bfP[j] = __builtin_bit_cast(_Float16, (unsigned short)(w >> 16));
        }
        v8f cT = {};
        cT = __builtin_amdgcn_wmma_f32_16x16x32_f16(
                 false, afrag, false, bfT, (short)0, cT, false, false);
        v8f cP = {};
        cP = __builtin_amdgcn_wmma_f32_16x16x32_f16(
                 false, afrag, false, bfP, (short)0, cP, false, false);

        if (g == 0) {  // D rows 0..2 (gx,gy,gz) live in lanes 0-15, c[0..2]
            float gxt = cT[0], gyt = cT[1], gzt = cT[2];
            float gxp = cP[0], gyp = cP[1], gzp = cP[2];
            float sqt = gxt * gxt + gyt * gyt + gzt * gzt;
            float sqp = gxp * gxp + gyp * gyp + gzp * gzp;
            int ci = base + ((z0 + zl) * VOL + (y0 + yl)) * VOL + (x0 + rn);
            if (mode == 0) {
                float m = mask[ci];
                accA += m;
                accB += __builtin_amdgcn_sqrtf(sqt + EPS_F) * m;
                accC += __builtin_amdgcn_sqrtf(sqp + EPS_F) * m;
            } else {
                float dent = __builtin_amdgcn_sqrtf(sqt + aet * aet + EPS_F);
                float denp = __builtin_amdgcn_sqrtf(sqp + aep * aep + EPS_F);
                float inner = (gxt * gxp + gyt * gyp + gzt * gzp) *
                              __builtin_amdgcn_rcpf(dent * denp);
                accA += inner * inner;
            }
        }
    }

    // Block reduction: LDS float atomics (ds_add_f32), then one global atomic.
    __hip_atomic_fetch_add(&red[0], accA, __ATOMIC_RELAXED, __HIP_MEMORY_SCOPE_WORKGROUP);
    if (mode == 0) {
        __hip_atomic_fetch_add(&red[1], accB, __ATOMIC_RELAXED, __HIP_MEMORY_SCOPE_WORKGROUP);
        __hip_atomic_fetch_add(&red[2], accC, __ATOMIC_RELAXED, __HIP_MEMORY_SCOPE_WORKGROUP);
    }
    __syncthreads();
    if (tid == 0) {
        if (mode == 0) {
            __hip_atomic_fetch_add(&ws[0 + b], red[0], __ATOMIC_RELAXED, __HIP_MEMORY_SCOPE_AGENT);
            __hip_atomic_fetch_add(&ws[2 + b], red[1], __ATOMIC_RELAXED, __HIP_MEMORY_SCOPE_AGENT);
            __hip_atomic_fetch_add(&ws[4 + b], red[2], __ATOMIC_RELAXED, __HIP_MEMORY_SCOPE_AGENT);
        } else {
            __hip_atomic_fetch_add(&ws[6], red[0], __ATOMIC_RELAXED, __HIP_MEMORY_SCOPE_AGENT);
        }
    }
}

__global__ void finalize_kernel(const float* __restrict__ ws, float* __restrict__ out) {
    if (threadIdx.x == 0) out[0] = 1.0f - ws[6] / NVOX;
}

extern "C" void kernel_launch(void* const* d_in, const int* in_sizes, int n_in,
                              void* d_out, int out_size, void* d_ws, size_t ws_size,
                              hipStream_t stream) {
    const float* pred = (const float*)d_in[0];
    const float* targ = (const float*)d_in[1];
    const float* mask = (const float*)d_in[2];
    float* ws  = (float*)d_ws;
    float* out = (float*)d_out;

    hipLaunchKernelGGL(init_ws_kernel, dim3(1), dim3(32), 0, stream, ws);

    dim3 grid(VOL / TILE, VOL / TILE, (VOL / TILE) * 2);  // 10 x 10 x (10*2 batches)
    dim3 block(256);
    // Pass 1: per-batch sums for auto_eps
    hipLaunchKernelGGL(sobel_pass_kernel, grid, block, 0, stream, pred, targ, mask, ws, 0);
    // Pass 2: normalized-gradient inner product accumulation (L2-resident re-read)
    hipLaunchKernelGGL(sobel_pass_kernel, grid, block, 0, stream, pred, targ, mask, ws, 1);

    hipLaunchKernelGGL(finalize_kernel, dim3(1), dim3(1), 0, stream, ws, out);
}
